// ColumnAttention_16243566313790
// MI455X (gfx1250) — matrix operand order repaired and checked by reading, not compiled, not run on hardware
//
#include <hip/hip_runtime.h>

// ---------------------------------------------------------------------------
// ColumnAttention for MI455X (gfx1250, wave32, WMMA bf16 16x16x32 + TDM)
//   out = softmax(query @ x^T + bias) @ x
//   B=64, N=576, D=1024, fp32 in/out, bf16 WMMA internally.
// Staging uses the Tensor Data Mover (tensor_load_to_lds, TENSORcnt) with
// double-buffered LDS so DMA of unit u+1 overlaps WMMA compute of unit u.
// ---------------------------------------------------------------------------

#define B_  64
#define N_  576
#define D_  1024

typedef __bf16 bf16;
typedef __bf16 v16bf __attribute__((ext_vector_type(16)));
typedef __bf16 v8bf  __attribute__((ext_vector_type(8)));
typedef float  v8f   __attribute__((ext_vector_type(8)));
typedef float  v4f   __attribute__((ext_vector_type(4)));
typedef unsigned int u32x4 __attribute__((ext_vector_type(4)));
typedef int          i32x8 __attribute__((ext_vector_type(8)));
typedef int          i32x4 __attribute__((ext_vector_type(4)));

// LDS pitches (elements); padded so 16-lane fragment reads hit distinct banks
#define QS_PITCH 1032   // 32 q-rows   x (1024 + 8) bf16
#define XS_PITCH 520    // 64 key-rows x (512 + 8)  bf16   (phase-1 unit)
#define XT_PITCH 40     // 512 d-rows  x (32 + 8)   bf16   (phase-2 unit)
#define SS_PITCH 584    // 32 rows x (576 + 8) f32  scores
#define PS_PITCH 584    // 32 rows x (576 + 8) bf16 probabilities

// byte offsets inside dynamic LDS
#define QS_OFF   0
#define X_OFF    66048          // 32*1032*2
#define XBUF_BYTES 66560        // 64*520*2 (>= 512*40*2 = 40960)
#define SS_OFF   199168         // X_OFF + 2*XBUF_BYTES
#define PS_OFF   273920         // + 32*584*4
#define SMEM_BYTES 311296       // + 32*584*2   (<= 320KB per WGP)

#if __has_builtin(__builtin_amdgcn_tensor_load_to_lds) && \
    __has_builtin(__builtin_amdgcn_s_wait_tensorcnt)
#define HAVE_TDM 1
#else
#define HAVE_TDM 0
#endif

// ---------------------------------------------------------------------------
// Kernel 0a: f32 -> bf16 elementwise (n divisible by 8)
// ---------------------------------------------------------------------------
__global__ __launch_bounds__(256) void cvt_bf16_kernel(
    const float* __restrict__ in, bf16* __restrict__ outp, int n) {
  int i = (blockIdx.x * 256 + threadIdx.x) * 8;
  if (i >= n) return;
  v4f a = *(const v4f*)(in + i);
  v4f b = *(const v4f*)(in + i + 4);
  v8bf o;
#pragma unroll
  for (int e = 0; e < 4; ++e) { o[e] = (bf16)a[e]; o[e + 4] = (bf16)b[e]; }
  *(v8bf*)(outp + i) = o;
}

// ---------------------------------------------------------------------------
// Kernel 0b: per-batch transpose  x[b][n][d] (f32) -> xt[b][d][n] (bf16)
// ---------------------------------------------------------------------------
__global__ __launch_bounds__(256) void transpose_kernel(
    const float* __restrict__ x, bf16* __restrict__ xt) {
  __shared__ float tile[64][68];
  const int b  = blockIdx.z;
  const int n0 = blockIdx.x * 64;
  const int d0 = blockIdx.y * 64;
  const int tid = threadIdx.x;
#pragma unroll
  for (int i = 0; i < 4; ++i) {
    int j = tid + i * 256;            // 1024 float4 per tile
    int r = j >> 4, c4 = (j & 15) * 4;
    v4f v = *(const v4f*)(x + ((size_t)(b * N_ + n0 + r)) * D_ + d0 + c4);
#pragma unroll
    for (int u = 0; u < 4; ++u) tile[r][c4 + u] = v[u];
  }
  __syncthreads();
#pragma unroll
  for (int i = 0; i < 2; ++i) {
    int j = tid + i * 256;            // 512 chunks of 8 bf16
    int c = j >> 3, r8 = (j & 7) * 8;
    v8bf o;
#pragma unroll
    for (int u = 0; u < 8; ++u) o[u] = (bf16)tile[r8 + u][c];
    *(v8bf*)(xt + ((size_t)(b * D_ + d0 + c)) * N_ + n0 + r8) = o;
  }
}

#if HAVE_TDM
// ---------------------------------------------------------------------------
// Issue one 2-D TDM tile load: global (row-strided) -> LDS (padded rows).
// D# layout per CDNA5 ISA 8.3-8.6.  Wave-uniform args -> SGPR groups.
// This toolchain exposes the 6-arg builtin:
//   (u32x4 g0, i32x8 g1, i32x4 g2, i32x4 g3, i32x8 gx, i32 cpol)
//   tile_w       : elements per row (contiguous, data_size=2B)
//   tile_rows    : number of rows
//   stride_elems : global stride between rows (elements)
//   pad_int_code : pad every (2<<code) DWORDs
//   pad_amt_code : pad by (code+1) DWORDs
// ---------------------------------------------------------------------------
__device__ __forceinline__ void tdm_load_2d(unsigned lds_byte, const void* gaddr,
                                            unsigned tile_w, unsigned tile_rows,
                                            unsigned long long stride_elems,
                                            unsigned pad_int_code,
                                            unsigned pad_amt_code) {
  unsigned long long ga = (unsigned long long)(uintptr_t)gaddr;
  u32x4 g0;
  g0[0] = 1u;                                        // count=1 (user mode)
  g0[1] = lds_byte;                                  // lds_addr
  g0[2] = (unsigned)(ga & 0xffffffffu);              // global_addr[31:0]
  g0[3] = (unsigned)((ga >> 32) & 0x1ffffffu)        // global_addr[56:32]
        | (2u << 30);                                // type = 2 ("image")
  i32x8 g1;
  g1[0] = (int)((1u << 16)                           // data_size = 2 bytes
              | (1u << 20)                           // pad_enable
              | (pad_int_code << 22)
              | (pad_amt_code << 25));
  const unsigned td0 = 0x40000000u;                  // huge tensor dims:
  const unsigned td1 = 0x40000000u;                  //   disable OOB clamping
  g1[1] = (int)((td0 & 0xffffu) << 16);              // tensor_dim0[15:0]
  g1[2] = (int)(((td0 >> 16) & 0xffffu) | ((td1 & 0xffffu) << 16));
  g1[3] = (int)(((td1 >> 16) & 0xffffu) | (tile_w << 16));   // tile_dim0
  g1[4] = (int)(tile_rows & 0xffffu);                // tile_dim1 (tile_dim2=0)
  g1[5] = (int)(stride_elems & 0xffffffffu);         // tensor_dim0_stride lo
  g1[6] = (int)((stride_elems >> 32) & 0xffffu);     // stride hi, dim1_stride=0
  g1[7] = 0;
  i32x4 g2 = {1, 1, 0, 0};                           // tensor_dim2/3 = 1
  i32x4 g3 = {0, (int)(1u << 16), 0, 0};             // tensor_dim4 = 1
  i32x8 gx = {0, 0, 0, 0, 0, 0, 0, 0};               // unused extended group
  __builtin_amdgcn_tensor_load_to_lds(g0, g1, g2, g3, gx, 0);
}
#endif

// ---------------------------------------------------------------------------
// Main fused attention kernel.
// Grid: (B, N/32). Block: 256 threads = 8 waves (wave32).
// Phase 1 units: (64 keys x 512 d), 18 units; 8 waves = 2 q-subtiles x 4 slots.
// Phase 2 units: (32 keys x 512 d), 36 units; each wave: 4 d-tiles x 2 q-tiles
//   per unit, 16 persistent f32 accumulator tiles (128 VGPRs).
// Double-buffered staging via TDM driven by wave 0.
// ---------------------------------------------------------------------------
__global__ __launch_bounds__(256) void attn_kernel(
    const bf16*  __restrict__ xb,    // [B][N][D] bf16
    const bf16*  __restrict__ xt,    // [B][D][N] bf16
    const bf16*  __restrict__ qb,    // [N][D]    bf16
    const float* __restrict__ bias,  // [N][N]
    float*       __restrict__ out)   // [B][N][D]
{
  extern __shared__ char smem[];
  bf16*  Qs = (bf16*)(smem + QS_OFF);
  bf16*  Xb0 = (bf16*)(smem + X_OFF);                // two staging buffers
  bf16*  Xb1 = (bf16*)(smem + X_OFF + XBUF_BYTES);
  float* Ss = (float*)(smem + SS_OFF);
  bf16*  Ps = (bf16*)(smem + PS_OFF);
#if HAVE_TDM
  const unsigned lds_base = __builtin_amdgcn_groupstaticsize() + X_OFF;
#endif

  const int b    = blockIdx.x;
  const int q0   = blockIdx.y * 32;
  const int tid  = threadIdx.x;
  const int lane = tid & 31;
  const int w    = tid >> 5;          // 0..7
  const int m    = lane & 15;
  const int hi   = lane >> 4;

#if HAVE_TDM
  // prologue: DMA phase-1 unit 0 while we stage Q
  if (w == 0) {
    tdm_load_2d(lds_base, xb + ((size_t)(b * N_) * D_),
                512, 64, (unsigned long long)D_, 7 /*256 DW*/, 3 /*4 DW*/);
  }
#endif

  // ---- stage Q tile: 32 rows x 1024 bf16 ----
  {
    const uint4* g = (const uint4*)(qb + (size_t)q0 * D_);
#pragma unroll
    for (int i = 0; i < 16; ++i) {
      int j = tid + i * 256;
      int r = j >> 7, off = j & 127;
      *(uint4*)(Qs + r * QS_PITCH + off * 8) = g[j];
    }
  }

  // =========================== Phase 1: scores ===========================
  const int qi   = w >> 2;            // q-subtile
  const int slot = w & 3;             // 16-key slot

  v8f acc = {};
  for (int u = 0; u < 18; ++u) {
    const int kc  = (u >> 1) * 64;
    const int dh  = u & 1;            // d-half: 0 -> d[0,512), 1 -> d[512,1024)
    const int par = u & 1;
    __syncthreads();                  // all waves done with buf[par] (unit u-2)
#if HAVE_TDM
    if (w == 0) {
      if (u + 1 < 18) {
        const int kc1 = ((u + 1) >> 1) * 64, dh1 = (u + 1) & 1;
        tdm_load_2d(lds_base + (unsigned)((par ^ 1) * XBUF_BYTES),
                    xb + ((size_t)(b * N_ + kc1) * D_ + dh1 * 512),
                    512, 64, (unsigned long long)D_, 7, 3);
        __builtin_amdgcn_s_wait_tensorcnt(1);   // unit u has landed
      } else {
        __builtin_amdgcn_s_wait_tensorcnt(0);
      }
    }
#else
    {   // cooperative fallback copy of unit u
      bf16* buf = par ? Xb1 : Xb0;
      const uint4* g = (const uint4*)(xb + ((size_t)(b * N_ + kc) * D_ + dh * 512));
#pragma unroll
      for (int i = 0; i < 16; ++i) {
        int j = tid + i * 256;        // 4096 uint4: r = j/64, off = j%64
        int r = j >> 6, off = j & 63;
        *(uint4*)(buf + r * XS_PITCH + off * 8) =
            *(const uint4*)((const bf16*)g + (size_t)r * D_ + off * 8);
      }
    }
#endif
    __syncthreads();                  // buf[par] now holds unit u

    const bf16* buf  = par ? Xb1 : Xb0;
    const bf16* arow = Qs + (qi * 16 + m) * QS_PITCH + dh * 512 + (hi ? 8 : 0);
    const bf16* brow = buf + (slot * 16 + m) * XS_PITCH + (hi ? 8 : 0);
#pragma unroll 4
    for (int dl = 0; dl < 512; dl += 32) {
      v8bf a0 = *(const v8bf*)(arow + dl);
      v8bf a1 = *(const v8bf*)(arow + dl + 16);
      v8bf b0 = *(const v8bf*)(brow + dl);
      v8bf b1 = *(const v8bf*)(brow + dl + 16);
      v16bf a, bm;
#pragma unroll
      for (int e = 0; e < 8; ++e) {
        a[e] = a0[e]; a[e + 8] = a1[e];
        bm[e] = b0[e]; bm[e + 8] = b1[e];
      }
      acc = __builtin_amdgcn_wmma_f32_16x16x32_bf16(
          false, a, false, bm, (short)0, acc, false, false);
    }
    if (dh == 1) {                    // full D accumulated: bias + store
#pragma unroll
      for (int g2 = 0; g2 < 8; ++g2) {
        int mr  = g2 + (hi ? 8 : 0);
        int col = kc + slot * 16 + m;
        float v = acc[g2] + bias[(size_t)(q0 + qi * 16 + mr) * N_ + col];
        Ss[(qi * 16 + mr) * SS_PITCH + col] = v;
      }
      acc = (v8f){};
    }
  }
  __syncthreads();

#if HAVE_TDM
  // prologue for phase 2: DMA unit 0 while softmax runs (Xbuf is idle now)
  if (w == 0) {
    tdm_load_2d(lds_base, xt + ((size_t)(b * D_) * N_),
                32, 512, (unsigned long long)N_, 3 /*16 DW*/, 3 /*4 DW*/);
  }
#endif

  // ============================ Softmax rows =============================
  {
    int r = tid >> 3;                 // 8 threads per row
    int s = tid & 7;
    float mx = -1e30f;
    for (int c = s; c < N_; c += 8) mx = fmaxf(mx, Ss[r * SS_PITCH + c]);
#pragma unroll
    for (int o = 4; o >= 1; o >>= 1) mx = fmaxf(mx, __shfl_xor(mx, o, 8));
    float sum = 0.f;
    for (int c = s; c < N_; c += 8) sum += __expf(Ss[r * SS_PITCH + c] - mx);
#pragma unroll
    for (int o = 4; o >= 1; o >>= 1) sum += __shfl_xor(sum, o, 8);
    float inv = 1.0f / sum;
    for (int c = s; c < N_; c += 8)
      Ps[r * PS_PITCH + c] = (bf16)(__expf(Ss[r * SS_PITCH + c] - mx) * inv);
  }

  // ========================= Phase 2: out = P @ x ========================
  v8f acc2[16];                       // [2 d-halves][4 d-tiles][2 q-subtiles]
#pragma unroll
  for (int t = 0; t < 16; ++t) acc2[t] = (v8f){};

  for (int u = 0; u < 36; ++u) {
    const int kc  = (u >> 1) * 32;
    const int dh  = u & 1;
    const int par = u & 1;
    __syncthreads();
#if HAVE_TDM
    if (w == 0) {
      if (u + 1 < 36) {
        const int kc1 = ((u + 1) >> 1) * 32, dh1 = (u + 1) & 1;
        tdm_load_2d(lds_base + (unsigned)((par ^ 1) * XBUF_BYTES),
                    xt + ((size_t)(b * D_ + dh1 * 512) * N_ + kc1),
                    32, 512, (unsigned long long)N_, 3, 3);
        __builtin_amdgcn_s_wait_tensorcnt(1);
      } else {
        __builtin_amdgcn_s_wait_tensorcnt(0);
      }
    }
#else
    {   // cooperative fallback copy of unit u
      bf16* buf = par ? Xb1 : Xb0;
#pragma unroll
      for (int i = 0; i < 8; ++i) {
        int j = tid + i * 256;        // 2048 uint4: dr = j/4, off = j%4
        int dr = j >> 2, off = j & 3;
        *(uint4*)(buf + dr * XT_PITCH + off * 8) =
            *(const uint4*)(xt + ((size_t)(b * D_ + dh * 512 + dr)) * N_ + kc + off * 8);
      }
    }
#endif
    __syncthreads();

    const bf16* buf = par ? Xb1 : Xb0;
    // A fragments: P[qi-tile 16 x 32 keys]
    v16bf afr[2];
#pragma unroll
    for (int q = 0; q < 2; ++q) {
      const bf16* ar = Ps + (q * 16 + m) * PS_PITCH + kc + (hi ? 8 : 0);
      v8bf a0 = *(const v8bf*)(ar);
      v8bf a1 = *(const v8bf*)(ar + 16);
#pragma unroll
      for (int e = 0; e < 8; ++e) { afr[q][e] = a0[e]; afr[q][e + 8] = a1[e]; }
    }
#pragma unroll
    for (int dt = 0; dt < 4; ++dt) {
      // B fragment: col n = local d, table-order keys; lanes>=16 -> keys+8
      const bf16* br = buf + ((w * 4 + dt) * 16 + m) * XT_PITCH + (hi ? 8 : 0);
      v8bf b0 = *(const v8bf*)(br);
      v8bf b1 = *(const v8bf*)(br + 16);
      v16bf bm;
#pragma unroll
      for (int e = 0; e < 8; ++e) { bm[e] = b0[e]; bm[e + 8] = b1[e]; }
      const int t = dh * 8 + dt * 2;
      acc2[t + 0] = __builtin_amdgcn_wmma_f32_16x16x32_bf16(
          false, afr[0], false, bm, (short)0, acc2[t + 0], false, false);
      acc2[t + 1] = __builtin_amdgcn_wmma_f32_16x16x32_bf16(
          false, afr[1], false, bm, (short)0, acc2[t + 1], false, false);
    }
  }

  // ---- write out tile: [32 q][1024 d] f32 ----
#pragma unroll
  for (int dh = 0; dh < 2; ++dh) {
#pragma unroll
    for (int dt = 0; dt < 4; ++dt) {
      int d0 = dh * 512 + (w * 4 + dt) * 16;
#pragma unroll
      for (int q = 0; q < 2; ++q) {
#pragma unroll
        for (int g2 = 0; g2 < 8; ++g2) {
          int mr = g2 + (hi ? 8 : 0);
          out[((size_t)(b * N_ + q0 + q * 16 + mr)) * D_ + d0 + m] =
              acc2[(dh * 8 + dt * 2) + q][g2];
        }
      }
    }
  }
}

// ---------------------------------------------------------------------------
// Host launcher.  Workspace layout (bf16):
//   [0]        x_bf16  [B][N][D]   75.5 MB
//   [B*N*D]    xT_bf16 [B][D][N]   75.5 MB
//   [2*B*N*D]  q_bf16  [N][D]       1.2 MB     (~152.3 MB of d_ws)
// ---------------------------------------------------------------------------
extern "C" void kernel_launch(void* const* d_in, const int* in_sizes, int n_in,
                              void* d_out, int out_size, void* d_ws, size_t ws_size,
                              hipStream_t stream) {
  const float* x     = (const float*)d_in[0];
  const float* query = (const float*)d_in[1];
  const float* bias  = (const float*)d_in[2];
  float* out = (float*)d_out;

  bf16* xb = (bf16*)d_ws;
  bf16* xt = xb + (size_t)B_ * N_ * D_;
  bf16* qb = xt + (size_t)B_ * N_ * D_;

  const int nx = B_ * N_ * D_;
  cvt_bf16_kernel<<<nx / (256 * 8), 256, 0, stream>>>(x, xb, nx);
  const int nq = N_ * D_;
  cvt_bf16_kernel<<<nq / (256 * 8), 256, 0, stream>>>(query, qb, nq);
  transpose_kernel<<<dim3(N_ / 64, D_ / 64, B_), 256, 0, stream>>>(x, xt);

  (void)hipFuncSetAttribute((const void*)attn_kernel,
                            hipFuncAttributeMaxDynamicSharedMemorySize,
                            SMEM_BYTES);
  attn_kernel<<<dim3(B_, N_ / 32), 256, SMEM_BYTES, stream>>>(xb, xt, qb, bias, out);
}
